// SelfAttention_12472585027711
// MI455X (gfx1250) — matrix operand order repaired and checked by reading, not compiled, not run on hardware
//
#include <hip/hip_runtime.h>
#include <hip/hip_bf16.h>
#include <cstddef>
#include <cstdint>

typedef __attribute__((ext_vector_type(16))) __bf16 bf16x16;
typedef __attribute__((ext_vector_type(8)))  float  f32x8;

#define B_   8
#define C_   512
#define CR_  32
#define N_   4096
#define LOG2E 1.44269504088896340736f

union AB16 { bf16x16 v; uint4 q[2]; };
union PK8 { uint4 q; __bf16 h[8]; };

static __device__ inline f32x8 wmma_bf16(const bf16x16& a, const bf16x16& b, const f32x8& c) {
    return __builtin_amdgcn_wmma_f32_16x16x32_bf16(false, a, false, b, (short)0, c, false, false);
}

// ---------------------------------------------------------------- cvt weights
__global__ __launch_bounds__(256)
void cvt_weights(const float* __restrict__ Wq, const float* __restrict__ Wk,
                 const float* __restrict__ Wv, const float* __restrict__ Wo,
                 __bf16* __restrict__ dst) {
    int i = blockIdx.x * 256 + threadIdx.x;
    if (i < 16384)       dst[i] = (__bf16)Wq[i];
    else if (i < 32768)  dst[i] = (__bf16)Wk[i - 16384];
    else if (i < 294912) dst[i] = (__bf16)Wv[i - 32768];
    else if (i < 557056) dst[i] = (__bf16)Wo[i - 294912];
}

// ------------------------------------------------------- QKV projection GEMMs
// grid (N/16, B), 256 threads (8 waves). Each wave: 64 v-channels (4 tiles).
// wave0 additionally computes q (2 cr-tiles), wave1 computes k.
__global__ __launch_bounds__(256)
void proj_kernel(const float* __restrict__ x,  const float* __restrict__ bq,
                 const float* __restrict__ bk, const float* __restrict__ bv,
                 const __bf16* __restrict__ Wqh, const __bf16* __restrict__ Wkh,
                 const __bf16* __restrict__ Wvh,
                 __bf16* __restrict__ qT, __bf16* __restrict__ kT,
                 __bf16* __restrict__ vv) {
    const int n0 = blockIdx.x * 16, b = blockIdx.y;
    const int lane = threadIdx.x & 31, wave = threadIdx.x >> 5;
    const int l16 = lane & 15, hi = lane >> 4;
    const int n = n0 + l16;
    const float* xb = x + (size_t)b * C_ * N_;

    f32x8 acc[4];
    #pragma unroll
    for (int t = 0; t < 4; ++t) {
        int cb = wave * 64 + t * 16 + 8 * hi;
        #pragma unroll
        for (int j = 0; j < 8; ++j) acc[t][j] = bv[cb + j];
    }
    f32x8 accqk[2] = {};
    if (wave == 0) {
        #pragma unroll
        for (int j = 0; j < 8; ++j) { accqk[0][j] = bq[8*hi + j]; accqk[1][j] = bq[16 + 8*hi + j]; }
    }
    if (wave == 1) {
        #pragma unroll
        for (int j = 0; j < 8; ++j) { accqk[0][j] = bk[8*hi + j]; accqk[1][j] = bk[16 + 8*hi + j]; }
    }

    for (int kc = 0; kc < 16; ++kc) {
        const int c0 = kc * 32;
        // B operand: x[c0+8*hi+(k&7)+16*(k>>3)][n] -> lane = column n, 16 K values
        bf16x16 bx;
        #pragma unroll
        for (int k2 = 0; k2 < 16; ++k2)
            bx[k2] = (__bf16)xb[(size_t)(c0 + 8 * hi + (k2 & 7) + ((k2 >> 3) << 4)) * N_ + n];
        #pragma unroll
        for (int t = 0; t < 4; ++t) {
            int row = wave * 64 + t * 16 + l16;
            const uint4* pa = (const uint4*)(Wvh + (size_t)row * C_ + c0 + 8 * hi);
            AB16 A; A.q[0] = pa[0]; A.q[1] = pa[2];   // K {0-7,16-23}/{8-15,24-31}
            acc[t] = wmma_bf16(A.v, bx, acc[t]);
        }
        if (wave < 2) {
            const __bf16* W = (wave == 0) ? Wqh : Wkh;
            #pragma unroll
            for (int t = 0; t < 2; ++t) {
                int row = t * 16 + l16;
                const uint4* pa = (const uint4*)(W + (size_t)row * C_ + c0 + 8 * hi);
                AB16 A; A.q[0] = pa[0]; A.q[1] = pa[2];
                accqk[t] = wmma_bf16(A.v, bx, accqk[t]);
            }
        }
    }
    // store v tiles: vv[b][c][n] bf16 (lane: n fixed, c = base+j)
    __bf16* vb = vv + (size_t)b * C_ * N_;
    #pragma unroll
    for (int t = 0; t < 4; ++t) {
        int cb = wave * 64 + t * 16 + 8 * hi;
        #pragma unroll
        for (int j = 0; j < 8; ++j) vb[(size_t)(cb + j) * N_ + n] = (__bf16)acc[t][j];
    }
    // store q/k transposed: [b][n][cr], packed 8 bf16 per b128 store
    if (wave < 2) {
        __bf16* dst = ((wave == 0) ? qT : kT) + ((size_t)b * N_ + n) * CR_;
        #pragma unroll
        for (int t = 0; t < 2; ++t) {
            PK8 pk;
            #pragma unroll
            for (int j = 0; j < 8; ++j) pk.h[j] = (__bf16)accqk[t][j];
            *(uint4*)(dst + t * 16 + 8 * hi) = pk.q;
        }
    }
}

// --------------------------------------------------------- flash attention
// grid (N/16, B), 256 threads. Two-pass softmax; probs staged in LDS.
__global__ __launch_bounds__(256)
void attn_kernel(const __bf16* __restrict__ qT, const __bf16* __restrict__ kT,
                 const __bf16* __restrict__ vv, __bf16* __restrict__ outp) {
    __shared__ float pstM[8][16];
    __shared__ float pstS[8][16];
    __shared__ float finM[16];
    __shared__ float finI[16];
    __shared__ __bf16 P[16][136];   // 128 probs + 8 pad (stride 272B, 16B aligned)

    const int n0 = blockIdx.x * 16, b = blockIdx.y;
    const int lane = threadIdx.x & 31, wave = threadIdx.x >> 5;
    const int l16 = lane & 15, hi = lane >> 4;
    const __bf16* qb = qT + (size_t)b * N_ * CR_;
    const __bf16* kb = kT + (size_t)b * N_ * CR_;
    const __bf16* vb = vv + (size_t)b * C_ * N_;

    AB16 Aq;   // persistent A operand: q rows n0..n0+15, K = cr
    { const uint4* p = (const uint4*)(qb + (size_t)(n0 + l16) * CR_ + 16 * hi);
      Aq.q[0] = p[0]; Aq.q[1] = p[1]; }

    // ---- pass 1: row max + sum-exp (wave w owns m-range [512w, 512w+512))
    float rm[8], rs[8];
    #pragma unroll
    for (int j = 0; j < 8; ++j) { rm[j] = -1e30f; rs[j] = 0.f; }
    for (int mc = 0; mc < 32; ++mc) {
        int m = wave * 512 + mc * 16 + l16;
        const uint4* p = (const uint4*)(kb + (size_t)m * CR_ + 16 * hi);
        AB16 Bk; Bk.q[0] = p[0]; Bk.q[1] = p[1];
        f32x8 z = {};
        f32x8 lg = wmma_bf16(Aq.v, Bk.v, z);
        #pragma unroll
        for (int j = 0; j < 8; ++j) {
            float l = lg[j] * LOG2E;
            float nm = fmaxf(rm[j], l);
            rs[j] = rs[j] * exp2f(rm[j] - nm) + exp2f(l - nm);
            rm[j] = nm;
        }
    }
    #pragma unroll
    for (int d = 1; d < 16; d <<= 1) {
        #pragma unroll
        for (int j = 0; j < 8; ++j) {
            float om = __shfl_xor(rm[j], d, 16);
            float os = __shfl_xor(rs[j], d, 16);
            float nm = fmaxf(rm[j], om);
            rs[j] = rs[j] * exp2f(rm[j] - nm) + os * exp2f(om - nm);
            rm[j] = nm;
        }
    }
    if (l16 == 0) {
        #pragma unroll
        for (int j = 0; j < 8; ++j) { pstM[wave][8*hi + j] = rm[j]; pstS[wave][8*hi + j] = rs[j]; }
    }
    __syncthreads();
    if (threadIdx.x < 16) {
        int r = threadIdx.x;
        float m = -1e30f;
        #pragma unroll
        for (int w = 0; w < 8; ++w) m = fmaxf(m, pstM[w][r]);
        float s = 0.f;
        #pragma unroll
        for (int w = 0; w < 8; ++w) s += pstS[w][r] * exp2f(pstM[w][r] - m);
        finM[r] = m; finI[r] = 1.0f / s;
    }
    __syncthreads();
    float rowM[8], rowI[8];
    #pragma unroll
    for (int j = 0; j < 8; ++j) { rowM[j] = finM[8*hi + j]; rowI[j] = finI[8*hi + j]; }

    // ---- pass 2: wave computes 16 prob columns per 128-m super-chunk,
    //              then all waves accumulate their 64 channels from LDS probs
    f32x8 acc[4] = {};
    const int cw = wave * 64;
    for (int sc = 0; sc < 32; ++sc) {
        {   // producer: one logits WMMA -> normalized probs into LDS
            int m = sc * 128 + wave * 16 + l16;
            const uint4* p = (const uint4*)(kb + (size_t)m * CR_ + 16 * hi);
            AB16 Bk; Bk.q[0] = p[0]; Bk.q[1] = p[1];
            f32x8 z = {};
            f32x8 lg = wmma_bf16(Aq.v, Bk.v, z);
            int mcol = wave * 16 + l16;
            #pragma unroll
            for (int j = 0; j < 8; ++j) {
                float pr = exp2f(lg[j] * LOG2E - rowM[j]) * rowI[j];
                P[8*hi + j][mcol] = (__bf16)pr;
            }
        }
        __syncthreads();
        #pragma unroll
        for (int kk = 0; kk < 4; ++kk) {
            const __bf16* pr = &P[l16][kk * 32 + 16 * hi];
            AB16 Ap; Ap.q[0] = *(const uint4*)pr; Ap.q[1] = *(const uint4*)(pr + 8);
            int mb = sc * 128 + kk * 32;
            #pragma unroll
            for (int t = 0; t < 4; ++t) {
                int c = cw + t * 16 + l16;
                const uint4* pv = (const uint4*)(vb + (size_t)c * N_ + mb + 16 * hi);
                AB16 Bv; Bv.q[0] = pv[0]; Bv.q[1] = pv[1];
                acc[t] = wmma_bf16(Ap.v, Bv.v, acc[t]);
            }
        }
        __syncthreads();
    }
    // store outp[b][c][n] bf16: lane c fixed, n = n0 + j + 8*hi consecutive
    __bf16* ob = outp + (size_t)b * C_ * N_;
    #pragma unroll
    for (int t = 0; t < 4; ++t) {
        int c = cw + t * 16 + l16;
        PK8 pk;
        #pragma unroll
        for (int j = 0; j < 8; ++j) pk.h[j] = (__bf16)acc[t][j];
        *(uint4*)(ob + (size_t)c * N_ + n0 + 8 * hi) = pk.q;
    }
}

// ------------------------------------- output projection + gamma*out + x
__global__ __launch_bounds__(256)
void oproj_kernel(const float* __restrict__ x, const float* __restrict__ bo,
                  const float* __restrict__ gamma, const __bf16* __restrict__ Woh,
                  const __bf16* __restrict__ outp, float* __restrict__ out) {
    const int n0 = blockIdx.x * 16, b = blockIdx.y;
    const int lane = threadIdx.x & 31, wave = threadIdx.x >> 5;
    const int l16 = lane & 15, hi = lane >> 4;
    const int n = n0 + l16;
    const __bf16* ib = outp + (size_t)b * C_ * N_;

    f32x8 acc[4];
    #pragma unroll
    for (int t = 0; t < 4; ++t) {
        int obch = wave * 64 + t * 16 + 8 * hi;
        #pragma unroll
        for (int j = 0; j < 8; ++j) acc[t][j] = bo[obch + j];
    }
    for (int kc = 0; kc < 16; ++kc) {
        int c0 = kc * 32;
        bf16x16 Bx;
        #pragma unroll
        for (int k2 = 0; k2 < 16; ++k2)
            Bx[k2] = ib[(size_t)(c0 + 8 * hi + (k2 & 7) + ((k2 >> 3) << 4)) * N_ + n];
        #pragma unroll
        for (int t = 0; t < 4; ++t) {
            int row = wave * 64 + t * 16 + l16;
            const uint4* pa = (const uint4*)(Woh + (size_t)row * C_ + c0 + 8 * hi);
            AB16 A; A.q[0] = pa[0]; A.q[1] = pa[2];
            acc[t] = wmma_bf16(A.v, Bx, acc[t]);
        }
    }
    float g = gamma[0];
    const float* xb = x + (size_t)b * C_ * N_;
    float* yb = out + (size_t)b * C_ * N_;
    #pragma unroll
    for (int t = 0; t < 4; ++t) {
        int cb = wave * 64 + t * 16 + 8 * hi;
        #pragma unroll
        for (int j = 0; j < 8; ++j) {
            size_t idx = (size_t)(cb + j) * N_ + n;
            yb[idx] = g * acc[t][j] + xb[idx];
        }
    }
}

// ---------------------------------------------------------------- launcher
extern "C" void kernel_launch(void* const* d_in, const int* in_sizes, int n_in,
                              void* d_out, int out_size, void* d_ws, size_t ws_size,
                              hipStream_t stream) {
    const float* x  = (const float*)d_in[0];
    const float* Wq = (const float*)d_in[1];
    const float* bq = (const float*)d_in[2];
    const float* Wk = (const float*)d_in[3];
    const float* bk = (const float*)d_in[4];
    const float* Wv = (const float*)d_in[5];
    const float* bv = (const float*)d_in[6];
    const float* Wo = (const float*)d_in[7];
    const float* bo = (const float*)d_in[8];
    const float* gm = (const float*)d_in[9];

    __bf16* Wqh = (__bf16*)d_ws;
    __bf16* Wkh = Wqh + (size_t)CR_ * C_;
    __bf16* Wvh = Wkh + (size_t)CR_ * C_;
    __bf16* Woh = Wvh + (size_t)C_ * C_;
    __bf16* qT  = Woh + (size_t)C_ * C_;
    __bf16* kT  = qT  + (size_t)B_ * N_ * CR_;
    __bf16* vv  = kT  + (size_t)B_ * N_ * CR_;
    __bf16* op  = vv  + (size_t)B_ * C_ * N_;

    cvt_weights<<<(557056 + 255) / 256, 256, 0, stream>>>(Wq, Wk, Wv, Wo, Wqh);
    dim3 grid(N_ / 16, B_);
    proj_kernel<<<grid, 256, 0, stream>>>(x, bq, bk, bv, Wqh, Wkh, Wvh, qT, kT, vv);
    attn_kernel<<<grid, 256, 0, stream>>>(qT, kT, vv, op);
    oproj_kernel<<<grid, 256, 0, stream>>>(x, bo, gm, Woh, op, (float*)d_out);
}
